// Transducer_ASR_13804024890033
// MI455X (gfx1250) — matrix operand order repaired and checked by reading, not compiled, not run on hardware
//
#include <hip/hip_runtime.h>
#include <hip/hip_bf16.h>
#include <math.h>

// ---------------- problem constants (match reference) ----------------
#define NB    8
#define TT    300
#define UU    60
#define U1    61
#define EE    320     // E_ENC == E_DEC
#define NHID  512
#define VOCAB 128
#define NP    (NB*TT*U1)      // 146400 joint positions (divisible by 16)
#define TU1   (TT*U1)         // 18300

typedef __attribute__((ext_vector_type(16))) __bf16 v16bf;
typedef __attribute__((ext_vector_type(8)))  float  v8f;

static __device__ __forceinline__ __bf16 to_bf16(float x) { return (__bf16)x; }

// =====================================================================
// Kernel 0: pack a row-major fp32 weight [K,N] into WMMA B-fragment
// order (bf16): [ktile][ntile][lane][16 contiguous bf16].
// B (32x16, 16-bit) layout: lanes 0-15 hold col=lane, K = kbase+{0..7, 16..23};
// lanes 16-31 hold col=lane-16, K = kbase+{8..15, 24..31}.
// =====================================================================
__global__ void pack_w_kernel(const float* __restrict__ W, int K, int N,
                              __bf16* __restrict__ out) {
    int id = blockIdx.x * blockDim.x + threadIdx.x;
    int ntiles = N >> 4, ktiles = K >> 5;
    int total = ktiles * ntiles * 32;
    if (id >= total) return;
    int lane = id & 31;
    int tmp  = id >> 5;
    int nt   = tmp % ntiles;
    int kt   = tmp / ntiles;
    int col   = nt * 16 + (lane & 15);
    int kbase = kt * 32 + (lane >> 4) * 8;
    __bf16* o = out + (size_t)id * 16;
#pragma unroll
    for (int j = 0; j < 16; ++j)
        o[j] = to_bf16(W[(size_t)(kbase + j + (j >= 8 ? 8 : 0)) * N + col]);
}

// =====================================================================
// A-fragment (16x32 bf16) built from one fp32 row.
// lanes 0-15 : row = lane,    K = k+{0..7, 16..23}
// lanes 16-31: row = lane-16, K = k+{8..15, 24..31}
// =====================================================================
static __device__ __forceinline__ v16bf load_a_frag(const float* __restrict__ row,
                                                    int k, int half) {
    const float* p0 = row + k + half * 8;
    const float* p1 = row + k + 16 + half * 8;
    v16bf a;
#pragma unroll
    for (int j = 0; j < 8; ++j) a[j]     = to_bf16(p0[j]);
#pragma unroll
    for (int j = 0; j < 8; ++j) a[8 + j] = to_bf16(p1[j]);
    return a;
}

// =====================================================================
// Kernel 1/2: C[M,512] = A[M,320] x Wpack(ktiles kt0..kt0+9)  (fp32 out)
// One wave per 16x16 output tile, 8 waves per block.
// =====================================================================
__global__ void gemm_joint_kernel(const float* __restrict__ A, int M,
                                  const __bf16* __restrict__ Wpack, int kt0,
                                  float* __restrict__ C) {
    const int ntiles = NHID / 16;                       // 32
    int wave = (blockIdx.x * blockDim.x + threadIdx.x) >> 5;
    int lane = threadIdx.x & 31;
    int mtiles = (M + 15) >> 4;
    if (wave >= mtiles * ntiles) return;                // wave-uniform
    int mt = wave / ntiles, nt = wave % ntiles;
    int half = lane >> 4;
    int row = mt * 16 + (lane & 15);
    if (row >= M) row = M - 1;                          // clamp (loads only)
    const float* arow = A + (size_t)row * EE;
    v8f acc = {};
#pragma unroll
    for (int kt = 0; kt < EE / 32; ++kt) {              // 10 k-steps
        v16bf a = load_a_frag(arow, kt * 32, half);
        const v16bf* bp = (const v16bf*)Wpack +
                          ((size_t)(kt0 + kt) * ntiles + nt) * 32 + lane;
        v16bf b = *bp;
        acc = __builtin_amdgcn_wmma_f32_16x16x32_bf16(false, a, false, b,
                                                      (short)0, acc, false, false);
    }
    int col = nt * 16 + (lane & 15);
#pragma unroll
    for (int r = 0; r < 8; ++r) {
        int m = mt * 16 + r + 8 * half;                 // C-frag row mapping
        if (m < M) C[(size_t)m * NHID + col] = acc[r];
    }
}

// =====================================================================
// Kernel 3: fused joint/tanh/vocab-GEMM/log-softmax reduction.
// Each wave owns 16 consecutive flattened lattice positions and the full
// 128-vocab output (8 f32 accumulators). Wo fragments are staged through
// LDS in 4 phases (4 k-steps each) so all 8 waves of the block share one
// global fetch; next phase is prefetched during compute.
// =====================================================================
static __device__ __forceinline__ float pick_acc(const v8f* acc, int v, int r) {
    float a0 = acc[0][r], a1 = acc[1][r], a2 = acc[2][r], a3 = acc[3][r];
    float a4 = acc[4][r], a5 = acc[5][r], a6 = acc[6][r], a7 = acc[7][r];
    float lo  = (v & 1) ? ((v & 2) ? a3 : a1) : ((v & 2) ? a2 : a0);
    float hi  = (v & 1) ? ((v & 2) ? a7 : a5) : ((v & 2) ? a6 : a4);
    return (v & 4) ? hi : lo;
}

__global__ void joint_vocab_kernel(const float* __restrict__ henc,
                                   const float* __restrict__ hdec,
                                   const float* __restrict__ bj,
                                   const __bf16* __restrict__ wopack,
                                   const float* __restrict__ bo,
                                   const int*  __restrict__ targets,
                                   float* __restrict__ blank_lp,
                                   float* __restrict__ emit_lp) {
    // LDS: one phase of Wo fragments (4 ksteps * 8 vtiles * 32 lanes * 16 bf16)
    __shared__ __bf16 lwo[4 * 8 * 32 * 16];             // 32 KB
    __shared__ float  lbj[NHID];                        // 2 KB

    int wave = (blockIdx.x * blockDim.x + threadIdx.x) >> 5;
    int lane = threadIdx.x & 31;
    int p0 = wave * 16;
    int half = lane >> 4;
    int m    = lane & 15;
    // position owned by this lane's A-matrix row (clamped so tail waves
    // stay alive for the workgroup barriers; their stores are suppressed)
    int p = p0 + m;
    if (p >= NP) p = NP - 1;
    int n = p / TU1; int rem = p - n * TU1;
    int t = rem / U1; int u = rem - t * U1;
    const float* er = henc + (size_t)(n * TT + t) * NHID;
    const float* dr = hdec + (size_t)(n * U1 + u) * NHID;

    // stage b_joint once
    for (int i = threadIdx.x; i < NHID; i += 256) lbj[i] = bj[i];

    v8f acc[8];
#pragma unroll
    for (int v = 0; v < 8; ++v) acc[v] = (v8f){};

    for (int ph = 0; ph < 4; ++ph) {
        __syncthreads();                                // protect prior reads
        // cooperative stage of 32 KB of Wo fragments (2048 x 16B)
        const uint4* src = (const uint4*)wopack + (size_t)ph * 2048;
        uint4* dst = (uint4*)lwo;
        for (int i = threadIdx.x; i < 2048; i += 256) dst[i] = src[i];
        __syncthreads();
        if (ph < 3)                                     // warm next chunk in L2
            __builtin_prefetch((const char*)wopack + (size_t)(ph + 1) * 32768 +
                               (size_t)threadIdx.x * 128, 0, 1);
#pragma unroll
        for (int kt8 = 0; kt8 < 4; ++kt8) {
            int kt = ph * 4 + kt8;
            int o0 = kt * 32 + half * 8;
            int o1 = kt * 32 + 16 + half * 8;
            v16bf a;
#pragma unroll
            for (int j = 0; j < 8; ++j)
                a[j]     = to_bf16(tanhf(er[o0 + j] + dr[o0 + j] + lbj[o0 + j]));
#pragma unroll
            for (int j = 0; j < 8; ++j)
                a[8 + j] = to_bf16(tanhf(er[o1 + j] + dr[o1 + j] + lbj[o1 + j]));
            const v16bf* bbase = (const v16bf*)lwo + (size_t)kt8 * 8 * 32 + lane;
#pragma unroll
            for (int vt = 0; vt < 8; ++vt) {
                v16bf b = bbase[vt * 32];
                acc[vt] = __builtin_amdgcn_wmma_f32_16x16x32_bf16(false, a, false, b,
                                                     (short)0, acc[vt], false, false);
            }
        }
    }
    // + output bias (column = vt*16 + lane%16)
#pragma unroll
    for (int vt = 0; vt < 8; ++vt) {
        float bov = bo[vt * 16 + (lane & 15)];
#pragma unroll
        for (int r = 0; r < 8; ++r) acc[vt][r] += bov;
    }
    // log-softmax stats per lattice row; row M = r + 8*half, p = p0 + M
#pragma unroll
    for (int r = 0; r < 8; ++r) {
        float mx = acc[0][r];
#pragma unroll
        for (int vt = 1; vt < 8; ++vt) mx = fmaxf(mx, acc[vt][r]);
#pragma unroll
        for (int s = 1; s < 16; s <<= 1) mx = fmaxf(mx, __shfl_xor(mx, s, 32));
        float sm = 0.f;
#pragma unroll
        for (int vt = 0; vt < 8; ++vt) sm += expf(acc[vt][r] - mx);
#pragma unroll
        for (int s = 1; s < 16; s <<= 1) sm += __shfl_xor(sm, s, 32);
        float lse = mx + logf(sm);

        int pr = p0 + r + 8 * half;
        bool prv = pr < NP;
        int prc = prv ? pr : NP - 1;
        int nn = prc / TU1; int rr = prc - nn * TU1;
        int tt = rr / U1;  int uu = rr - tt * U1;

        // blank (vocab 0) lives in acc[0][r] on lane (half*16 + 0)
        float blogit = __shfl(acc[0][r], half * 16, 32);
        if (prv && (lane & 15) == 0)
            blank_lp[prc] = blogit - lse;

        if (uu < UU) {
            int c  = targets[nn * UU + uu];             // uniform within half
            int vs = (c >> 4) & 7, cl = c & 15;
            float val = pick_acc(acc, vs, r);
            float elogit = __shfl(val, half * 16 + cl, 32);
            if (prv && (lane & 15) == 0)
                emit_lp[(size_t)(nn * TT + tt) * UU + uu] = elogit - lse;
        }
    }
}

// =====================================================================
// Kernel 4: RNNT forward DP via anti-diagonal wavefront.
// One wave32 per batch (8 waves, single block); alpha row in LDS;
// T+U ~ 360 serial steps with 61-way lane parallelism.
// =====================================================================
static __device__ __forceinline__ float logaddexpf_(float x, float y) {
    float mx = fmaxf(x, y), mn = fminf(x, y);
    if (mx == -INFINITY) return -INFINITY;
    return mx + log1pf(expf(mn - mx));
}

__global__ void rnnt_dp_kernel(const float* __restrict__ blank_lp,
                               const float* __restrict__ emit_lp,
                               const int* __restrict__ inputs_len,
                               const int* __restrict__ targets_len,
                               float* __restrict__ out) {
    __shared__ float A[NB][64];
    __shared__ float lossS[NB];
    int n    = threadIdx.x >> 5;
    int lane = threadIdx.x & 31;
    int lt = inputs_len[n];
    int lu = targets_len[n];
    const float* B = blank_lp + (size_t)n * TT * U1;
    const float* E = emit_lp  + (size_t)n * TT * UU;
    float* a = A[n];
    a[lane]      = -INFINITY;
    a[lane + 32] = -INFINITY;

    int u1 = lane, u2 = lane + 32;
    for (int d = 0; d <= lt - 1 + lu; ++d) {
        // ---- read phase (old alpha) ----
        int t1 = d - u1, t2 = d - u2;
        bool ok1 = (u1 <= lu) && (t1 >= 0) && (t1 < lt);
        bool ok2 = (u2 <= lu) && (t2 >= 0) && (t2 < lt);
        float v1 = -INFINITY, v2 = -INFINITY;
        if (ok1) {
            float fb = (t1 > 0) ? a[u1] + B[(size_t)(t1 - 1) * U1 + u1] : -INFINITY;
            float fe = (u1 > 0) ? a[u1 - 1] + E[(size_t)t1 * UU + (u1 - 1)] : -INFINITY;
            v1 = (t1 == 0 && u1 == 0) ? 0.0f : logaddexpf_(fb, fe);
        }
        if (ok2) {
            float fb = (t2 > 0) ? a[u2] + B[(size_t)(t2 - 1) * U1 + u2] : -INFINITY;
            float fe = a[u2 - 1] + E[(size_t)t2 * UU + (u2 - 1)];       // u2 >= 32 > 0
            v2 = logaddexpf_(fb, fe);
        }
        // ---- write phase ----
        if (ok1) a[u1] = v1;
        if (ok2) a[u2] = v2;
    }
    if (lane == 0)
        lossS[n] = -(a[lu] + B[(size_t)(lt - 1) * U1 + lu]);
    __syncthreads();
    if (threadIdx.x == 0) {
        float s = 0.f;
#pragma unroll
        for (int i = 0; i < NB; ++i) s += lossS[i];
        out[0] = s / (float)NB;
    }
}

// =====================================================================
// Host-side orchestration
// =====================================================================
static inline size_t align256(size_t x) { return (x + 255) & ~(size_t)255; }

extern "C" void kernel_launch(void* const* d_in, const int* in_sizes, int n_in,
                              void* d_out, int out_size, void* d_ws, size_t ws_size,
                              hipStream_t stream) {
    (void)in_sizes; (void)n_in; (void)out_size; (void)ws_size;
    const float* enc   = (const float*)d_in[0];   // [8,300,320]
    const float* dec   = (const float*)d_in[1];   // [8,61,320]
    const int*   tgt   = (const int*)  d_in[2];   // [8,60]
    const int*   ilen  = (const int*)  d_in[3];   // [8]
    const int*   tlen  = (const int*)  d_in[4];   // [8]
    const float* Wj    = (const float*)d_in[5];   // [640,512]
    const float* bj    = (const float*)d_in[6];   // [512]
    const float* Wo    = (const float*)d_in[7];   // [512,128]
    const float* bo    = (const float*)d_in[8];   // [128]
    float* out = (float*)d_out;

    // workspace carve-up
    char* ws = (char*)d_ws;
    size_t off = 0;
    float*  henc   = (float*) (ws + off); off = align256(off + sizeof(float)  * (size_t)NB * TT * NHID);
    float*  hdec   = (float*) (ws + off); off = align256(off + sizeof(float)  * (size_t)NB * U1 * NHID);
    __bf16* wjpack = (__bf16*)(ws + off); off = align256(off + sizeof(__bf16) * (size_t)(2 * EE) * NHID);
    __bf16* wopack = (__bf16*)(ws + off); off = align256(off + sizeof(__bf16) * (size_t)NHID * VOCAB);
    float*  blank  = (float*) (ws + off); off = align256(off + sizeof(float)  * (size_t)NP);
    float*  emit   = (float*) (ws + off); off = align256(off + sizeof(float)  * (size_t)NB * TT * UU);

    // 0) pack weights into WMMA B-fragment order (bf16)
    {
        int tot = (2 * EE / 32) * (NHID / 16) * 32;     // 20*32*32 = 20480
        pack_w_kernel<<<(tot + 255) / 256, 256, 0, stream>>>(Wj, 2 * EE, NHID, wjpack);
    }
    {
        int tot = (NHID / 32) * (VOCAB / 16) * 32;      // 16*8*32 = 4096
        pack_w_kernel<<<(tot + 255) / 256, 256, 0, stream>>>(Wo, NHID, VOCAB, wopack);
    }

    // 1) henc = enc @ Wj[:320]   (M = 2400 -> 150*32 = 4800 wave tiles)
    {
        int M = NB * TT;
        int waves = ((M + 15) / 16) * (NHID / 16);
        gemm_joint_kernel<<<(waves + 7) / 8, 256, 0, stream>>>(enc, M, wjpack, 0, henc);
    }
    // 2) hdec = dec @ Wj[320:]   (M = 488 -> 31*32 = 992 wave tiles)
    {
        int M = NB * U1;
        int waves = ((M + 15) / 16) * (NHID / 16);
        gemm_joint_kernel<<<(waves + 7) / 8, 256, 0, stream>>>(dec, M, wjpack, EE / 32, hdec);
    }
    // 3) fused tanh-joint + vocab GEMM (LDS-staged Wo) + log-softmax gather
    {
        int waves = NP / 16;                            // 9150
        joint_vocab_kernel<<<(waves + 7) / 8, 256, 0, stream>>>(
            henc, hdec, bj, wopack, bo, tgt, blank, emit);
    }
    // 4) wavefront DP + mean loss (single block, deterministic)
    rnnt_dp_kernel<<<1, NB * 32, 0, stream>>>(blank, emit, ilen, tlen, out);
}
